// SpatialGNN_80083960201605
// MI455X (gfx1250) — compile-verified, matrix operand order checked by reading, not run on hardware
//
#include <hip/hip_runtime.h>
#include <hip/hip_bf16.h>

// ---------------- problem constants ----------------
#define NN    50000
#define EE    400000
#define ETOT  (EE + NN)          // edges + self loops
#define FIN   300
#define KPAD1 320                // 300 padded to multiple of 32
#define HID   128
#define HEADS 4
#define FWID  512                // HEADS*HID, also row stride of all activation buffers
#define NCLS  20
#define NEG_SLOPE 0.2f
#define MTILES (NN / 16)         // 3125

typedef __attribute__((ext_vector_type(16))) __bf16 v16bf;
typedef __attribute__((ext_vector_type(8)))  float  v8f;
typedef __attribute__((ext_vector_type(4)))  unsigned int v4u;

// ---------------- helpers ----------------
__device__ __forceinline__ unsigned short f2bf(float f) {
    unsigned u = __float_as_uint(f);
    unsigned r = u + 0x7FFFu + ((u >> 16) & 1u);   // round-to-nearest-even
    return (unsigned short)(r >> 16);
}
__device__ __forceinline__ unsigned fenc(float f) {   // order-preserving float->uint
    unsigned u = __float_as_uint(f);
    return (u & 0x80000000u) ? ~u : (u | 0x80000000u);
}
__device__ __forceinline__ float fdec(unsigned u) {
    return (u & 0x80000000u) ? __uint_as_float(u ^ 0x80000000u) : __uint_as_float(~u);
}

// ---------------- cast x (fp32, N x 300) -> bf16 (N x 512 stride, K padded to 320) --------
__global__ void cast_x_bf16(const float* __restrict__ x, unsigned short* __restrict__ xb) {
    long i = (long)blockIdx.x * blockDim.x + threadIdx.x;
    if (i >= (long)NN * KPAD1) return;
    int n = (int)(i / KPAD1), k = (int)(i % KPAD1);
    xb[(long)n * FWID + k] = (k < FIN) ? f2bf(x[(long)n * FIN + k]) : (unsigned short)0;
}

// ---------------- pack W (Kin x NoutOrig fp32) into WMMA-B fragment order -----------------
// layout: [ntile][ks][lane][e(0..15)] bf16, matching ISA 7.12.2 16-bit B layout
__global__ void pack_w(const float* __restrict__ W, unsigned short* __restrict__ wf,
                       int Kin, int NoutOrig, int Ksteps, int total) {
    int id = blockIdx.x * blockDim.x + threadIdx.x;
    if (id >= total) return;
    int e    = id & 15;
    int lane = (id >> 4) & 31;
    int rem  = id >> 9;            // ntile*Ksteps + ks
    int ks   = rem % Ksteps;
    int nt   = rem / Ksteps;
    int ncol  = nt * 16 + (lane & 15);
    int kbase = (lane >> 4) * 8;
    int j = e >> 1;
    int koff = ((j < 4) ? (kbase + 2 * j) : (16 + kbase + 2 * (j - 4))) + (e & 1);
    int k = ks * 32 + koff;
    float v = (k < Kin && ncol < NoutOrig) ? W[(long)k * NoutOrig + ncol] : 0.0f;
    wf[id] = f2bf(v);
}

// ---------------- bf16 WMMA GEMM ----------------
// one block per 16-row M tile: A tile (16 x Kpad bf16, <=16KB) staged in LDS once,
// each wave computes NT consecutive 16x16 N tiles (compile-time NT -> static acc regs).
template <int NT, int KSTEPS>
__global__ void gemm_bf16_wmma(const unsigned short* __restrict__ xb,
                               const unsigned short* __restrict__ wf,
                               float* __restrict__ xp,
                               int Ntiles) {
    constexpr int VPR = (KSTEPS * 32) >> 3;                // v4u per LDS row (40 or 64)
    __shared__ __align__(16) unsigned short lA[16 * KSTEPS * 32];

    const int mtile = blockIdx.x;

    // ---- stage A tile into LDS (global row stride is FWID bf16 = 64 v4u) ----
    const v4u* gA = (const v4u*)xb;
    v4u*       sA = (v4u*)lA;
    for (int i = threadIdx.x; i < 16 * VPR; i += 256) {
        int r = i / VPR, c = i % VPR;
        sA[r * VPR + c] = gA[(long)(mtile * 16 + r) * (FWID / 8) + c];
    }
    __syncthreads();

    const int wv   = threadIdx.x >> 5;
    const int lane = threadIdx.x & 31;
    const int nt0  = wv * NT;
    if (nt0 < Ntiles) {                                    // wave-uniform guard
        const int mloc  = lane & 15;
        const int kbase = (lane >> 4) * 8;
        const v4u* aRow = sA + mloc * VPR;                 // LDS reads (ds_load_b128)
        const v4u* bBase = (const v4u*)wf + ((long)nt0 * KSTEPS * 32 + lane) * 2;

        v8f acc[NT];
#pragma unroll
        for (int nt = 0; nt < NT; ++nt) acc[nt] = (v8f){};

        for (int ks = 0; ks < KSTEPS; ++ks) {
            union { v16bf v; v4u q[2]; } A;
            A.q[0] = aRow[ks * 4 + (kbase >> 3)];          // K = ks*32 + kbase .. +7
            A.q[1] = aRow[ks * 4 + 2 + (kbase >> 3)];      // K = ks*32 + 16 + kbase .. +7
#pragma unroll
            for (int nt = 0; nt < NT; ++nt) {
                union { v16bf v; v4u q[2]; } B;
                const v4u* bp = bBase + ((long)nt * KSTEPS + ks) * 64;
                B.q[0] = bp[0];
                B.q[1] = bp[1];
                acc[nt] = __builtin_amdgcn_wmma_f32_16x16x32_bf16(
                    false, A.v, false, B.v, (short)0, acc[nt], false, false);
            }
        }

        const int mrow0 = mtile * 16 + (lane >> 4) * 8;
#pragma unroll
        for (int nt = 0; nt < NT; ++nt) {
            int ncol = (nt0 + nt) * 16 + (lane & 15);
#pragma unroll
            for (int j = 0; j < 8; ++j)
                xp[(long)(mrow0 + j) * FWID + ncol] = acc[nt][j];
        }
    }
}

// ---------------- attention scores: a_s/a_d[n,h] = sum_c xp[n,h*C+c]*att[h,c] ------------
__global__ void attn_scores(const float* __restrict__ xp,
                            const float* __restrict__ att_src,
                            const float* __restrict__ att_dst,
                            float* __restrict__ a_s, float* __restrict__ a_d,
                            int H, int C) {
    int gw   = (int)((blockIdx.x * (long)blockDim.x + threadIdx.x) >> 5);
    int lane = threadIdx.x & 31;
    if (gw >= NN * H) return;
    int n = gw / H, h = gw % H;
    const float* row = xp + (long)n * FWID + h * C;
    float s = 0.f, d = 0.f;
    for (int c = lane; c < C; c += 32) {
        float v = row[c];
        s += v * att_src[h * C + c];
        d += v * att_dst[h * C + c];
    }
    for (int off = 16; off; off >>= 1) {
        s += __shfl_xor(s, off, 32);
        d += __shfl_xor(d, off, 32);
    }
    if (lane == 0) { a_s[gw] = s; a_d[gw] = d; }
}

// ---------------- edge pass 1: e = leaky_relu(a_s[src]+a_d[dst]); segment max ------------
__global__ void edge_scores(const int* __restrict__ ei,
                            const float* __restrict__ a_s, const float* __restrict__ a_d,
                            float* __restrict__ e_edge, unsigned* __restrict__ m_u, int H) {
    long i = (long)blockIdx.x * blockDim.x + threadIdx.x;
    if (i >= (long)ETOT * H) return;
    int eidx = (int)(i / H), h = (int)(i % H);
    int s, d;
    if (eidx < EE) { s = ei[eidx]; d = ei[EE + eidx]; } else { s = d = eidx - EE; }
    float v = a_s[(long)s * H + h] + a_d[(long)d * H + h];
    v = (v > 0.f) ? v : NEG_SLOPE * v;
    e_edge[i] = v;
    atomicMax(m_u + (long)d * H + h, fenc(v));
}

// ---------------- decode ordered-uint max back to float (in place) -----------------------
__global__ void decode_max(unsigned* __restrict__ m_u, int total) {
    int i = blockIdx.x * blockDim.x + threadIdx.x;
    if (i >= total) return;
    ((float*)m_u)[i] = fdec(m_u[i]);
}

// ---------------- edge pass 2: ex = exp(e - m[dst]); segment sum -------------------------
__global__ void edge_exp(const int* __restrict__ ei, float* __restrict__ e_edge,
                         const float* __restrict__ m, float* __restrict__ denom, int H) {
    long i = (long)blockIdx.x * blockDim.x + threadIdx.x;
    if (i >= (long)ETOT * H) return;
    int eidx = (int)(i / H), h = (int)(i % H);
    int d = (eidx < EE) ? ei[EE + eidx] : (eidx - EE);
    float ex = __expf(e_edge[i] - m[(long)d * H + h]);
    e_edge[i] = ex;
    atomicAdd(denom + (long)d * H + h, ex);
}

// ---------------- edge pass 3: out[dst] += alpha * xp[src]  (warp per edge-head) ---------
__global__ void edge_aggregate(const int* __restrict__ ei,
                               const float* __restrict__ e_ex,
                               const float* __restrict__ denom,
                               const float* __restrict__ xp,
                               float* __restrict__ outb, int H, int C) {
    long gw  = (blockIdx.x * (long)blockDim.x + threadIdx.x) >> 5;
    int lane = threadIdx.x & 31;
    if (gw >= (long)ETOT * H) return;
    int eidx = (int)(gw / H), h = (int)(gw % H);
    int s, d;
    if (eidx < EE) { s = ei[eidx]; d = ei[EE + eidx]; } else { s = d = eidx - EE; }
    const float* xr   = xp + (long)s * FWID + h * C;
    float*       orow = outb + (long)d * FWID + h * C;
    __builtin_prefetch(xr + lane, 0, 1);            // global_prefetch: hide gather latency
    float alpha = e_ex[gw] / (denom[(long)d * H + h] + 1e-16f);
    for (int c = lane; c < C; c += 32)
        atomicAdd(orow + c, alpha * xr[c]);
}

// ---------------- epilogue (layers 1,2): bias + relu + cast to bf16 ----------------------
__global__ void bias_relu_cast(const float* __restrict__ outb, const float* __restrict__ b,
                               unsigned short* __restrict__ xb) {
    long i = (long)blockIdx.x * blockDim.x + threadIdx.x;
    if (i >= (long)NN * FWID) return;
    int c = (int)(i & (FWID - 1));
    float v = outb[i] + b[c];
    v = v > 0.f ? v : 0.f;
    xb[i] = f2bf(v);
}

// ---------------- final: log_softmax over 20 classes (warp per node) ---------------------
__global__ void final_logsoftmax(const float* __restrict__ outb, const float* __restrict__ b,
                                 float* __restrict__ out) {
    int gw   = (int)((blockIdx.x * (long)blockDim.x + threadIdx.x) >> 5);
    int lane = threadIdx.x & 31;
    if (gw >= NN) return;
    float v = (lane < NCLS) ? (outb[(long)gw * FWID + lane] + b[lane]) : -3.4e38f;
    float m = v;
    for (int off = 16; off; off >>= 1) m = fmaxf(m, __shfl_xor(m, off, 32));
    float ex = (lane < NCLS) ? __expf(v - m) : 0.f;
    float ssum = ex;
    for (int off = 16; off; off >>= 1) ssum += __shfl_xor(ssum, off, 32);
    if (lane < NCLS) out[(long)gw * NCLS + lane] = v - m - __logf(ssum);
}

// ---------------- host-side layer driver ----------------
struct LayerCfg { int Kin, Ksteps, NoutOrig, Ntiles, H, C; };

static inline int cdiv(long a, long b) { return (int)((a + b - 1) / b); }

static void run_gat_edges(hipStream_t stream,
                          const float* att_src, const float* att_dst, const int* ei,
                          float* xp, float* outb, float* a_s, float* a_d,
                          unsigned* m_u, float* denom, float* e_edge, int H, int C) {
    const int NH = NN * H;
    attn_scores<<<cdiv((long)NH * 32, 256), 256, 0, stream>>>(xp, att_src, att_dst,
                                                              a_s, a_d, H, C);
    long EH = (long)ETOT * H;
    edge_scores<<<cdiv(EH, 256), 256, 0, stream>>>(ei, a_s, a_d, e_edge, m_u, H);
    decode_max<<<cdiv(NH, 256), 256, 0, stream>>>(m_u, NH);
    edge_exp<<<cdiv(EH, 256), 256, 0, stream>>>(ei, e_edge, (const float*)m_u, denom, H);
    edge_aggregate<<<cdiv(EH * 32, 256), 256, 0, stream>>>(ei, e_edge, denom, xp, outb, H, C);
}

extern "C" void kernel_launch(void* const* d_in, const int* in_sizes, int n_in,
                              void* d_out, int out_size, void* d_ws, size_t ws_size,
                              hipStream_t stream) {
    const float* x   = (const float*)d_in[0];
    const int*   ei  = (const int*)d_in[1];            // [2, E]
    const float* W1  = (const float*)d_in[2];
    const float* as1 = (const float*)d_in[3];
    const float* ad1 = (const float*)d_in[4];
    const float* b1  = (const float*)d_in[5];
    const float* W2  = (const float*)d_in[6];
    const float* as2 = (const float*)d_in[7];
    const float* ad2 = (const float*)d_in[8];
    const float* b2  = (const float*)d_in[9];
    const float* W3  = (const float*)d_in[10];
    const float* as3 = (const float*)d_in[11];
    const float* ad3 = (const float*)d_in[12];
    const float* b3  = (const float*)d_in[13];
    float* out = (float*)d_out;

    // ---- carve workspace ----
    char* w = (char*)d_ws;
    auto take = [&](size_t bytes) { void* p = (void*)w; w += (bytes + 255) & ~(size_t)255; return p; };
    float*          xp     = (float*)take((size_t)NN * FWID * 4);
    float*          outb   = (float*)take((size_t)NN * FWID * 4);
    unsigned short* xb     = (unsigned short*)take((size_t)NN * FWID * 2);
    unsigned short* wfrag  = (unsigned short*)take((size_t)32 * 16 * 512 * 2);
    float*          a_s    = (float*)take((size_t)NN * HEADS * 4);
    float*          a_d    = (float*)take((size_t)NN * HEADS * 4);
    unsigned*       m_u    = (unsigned*)take((size_t)NN * HEADS * 4);
    float*          denom  = (float*)take((size_t)NN * HEADS * 4);
    float*          e_edge = (float*)take((size_t)ETOT * HEADS * 4);

    // ---- input cast to bf16 (K padded 300 -> 320, row stride 512) ----
    cast_x_bf16<<<cdiv((long)NN * KPAD1, 256), 256, 0, stream>>>(x, xb);

    // ================= layer 1: GAT(300 -> 4x128) =================
    {
        const int Ksteps = KPAD1 / 32, Ntiles = FWID / 16;
        hipMemsetAsync(m_u,   0, (size_t)NN * HEADS * 4, stream);
        hipMemsetAsync(denom, 0, (size_t)NN * HEADS * 4, stream);
        hipMemsetAsync(outb,  0, (size_t)NN * FWID * 4, stream);
        int packTotal = Ntiles * Ksteps * 512;
        pack_w<<<cdiv(packTotal, 256), 256, 0, stream>>>(W1, wfrag, FIN, FWID, Ksteps, packTotal);
        gemm_bf16_wmma<4, KPAD1 / 32><<<MTILES, 256, 0, stream>>>(xb, wfrag, xp, Ntiles);
        run_gat_edges(stream, as1, ad1, ei, xp, outb, a_s, a_d, m_u, denom, e_edge, HEADS, HID);
        bias_relu_cast<<<cdiv((long)NN * FWID, 256), 256, 0, stream>>>(outb, b1, xb);
    }

    // ================= layer 2: GAT(512 -> 4x128) =================
    {
        const int Ksteps = FWID / 32, Ntiles = FWID / 16;
        hipMemsetAsync(m_u,   0, (size_t)NN * HEADS * 4, stream);
        hipMemsetAsync(denom, 0, (size_t)NN * HEADS * 4, stream);
        hipMemsetAsync(outb,  0, (size_t)NN * FWID * 4, stream);
        int packTotal = Ntiles * Ksteps * 512;
        pack_w<<<cdiv(packTotal, 256), 256, 0, stream>>>(W2, wfrag, FWID, FWID, Ksteps, packTotal);
        gemm_bf16_wmma<4, FWID / 32><<<MTILES, 256, 0, stream>>>(xb, wfrag, xp, Ntiles);
        run_gat_edges(stream, as2, ad2, ei, xp, outb, a_s, a_d, m_u, denom, e_edge, HEADS, HID);
        bias_relu_cast<<<cdiv((long)NN * FWID, 256), 256, 0, stream>>>(outb, b2, xb);
    }

    // ================= layer 3: GAT(512 -> 20, 1 head) + log_softmax =================
    {
        const int Ksteps = FWID / 32, Ntiles = 2;
        hipMemsetAsync(m_u,   0, (size_t)NN * 4, stream);
        hipMemsetAsync(denom, 0, (size_t)NN * 4, stream);
        hipMemsetAsync(outb,  0, (size_t)NN * FWID * 4, stream);
        int packTotal = Ntiles * Ksteps * 512;
        pack_w<<<cdiv(packTotal, 256), 256, 0, stream>>>(W3, wfrag, FWID, NCLS, Ksteps, packTotal);
        gemm_bf16_wmma<1, FWID / 32><<<MTILES, 256, 0, stream>>>(xb, wfrag, xp, Ntiles);
        run_gat_edges(stream, as3, ad3, ei, xp, outb, a_s, a_d, m_u, denom, e_edge, 1, NCLS);
        final_logsoftmax<<<cdiv((long)NN * 32, 256), 256, 0, stream>>>(outb, b3, out);
    }
}
